// BlockDrop_24455543784015
// MI455X (gfx1250) — matrix operand order repaired and checked by reading, not compiled, not run on hardware
//
#include <hip/hip_runtime.h>
#include <stddef.h>

typedef __attribute__((ext_vector_type(16))) __bf16 v16bf;
typedef __attribute__((ext_vector_type(8)))  float  v8f;
typedef __attribute__((ext_vector_type(4)))  float  float4v;
typedef __attribute__((ext_vector_type(4)))  unsigned short ushort4v;
typedef __attribute__((ext_vector_type(8)))  unsigned short ushort8v;

__device__ __forceinline__ unsigned short f2bf_u16(float f) {
  unsigned u = __builtin_bit_cast(unsigned, f);
  u += 0x7FFFu + ((u >> 16) & 1u);           // round-to-nearest-even
  return (unsigned short)(u >> 16);
}
__device__ __forceinline__ __bf16 bf_from_u16(unsigned short u) {
  return __builtin_bit_cast(__bf16, u);
}

// ---------------- prep kernels ----------------
__global__ void fold_bn_kernel(const float* __restrict__ g, const float* __restrict__ b,
                               const float* __restrict__ m, const float* __restrict__ v,
                               float* __restrict__ scale, float* __restrict__ bias, int C) {
  int c = blockIdx.x * blockDim.x + threadIdx.x;
  if (c >= C) return;
  float s = g[c] * rsqrtf(v[c] + 1e-3f);
  scale[c] = s;
  bias[c]  = b[c] - m[c] * s;
}

// HWIO weights are [K=9*Cin, Cout] row-major; convert to bf16 TRANSPOSED: dst[co*Kpad + k],
// zero-padded K -> Kpad, so the GEMM-B tile can be staged/read contiguously.
__global__ void cvt_w_kernel(const float* __restrict__ src, unsigned short* __restrict__ dst,
                             int K, int Kpad, int Cout) {
  int i = blockIdx.x * blockDim.x + threadIdx.x;
  if (i >= Kpad * Cout) return;
  int co = i / Kpad;
  int k  = i - co * Kpad;
  dst[i] = (k < K) ? f2bf_u16(src[(size_t)k * Cout + co]) : (unsigned short)0;
}

// ---------------- implicit-GEMM conv3x3 + folded BN (+ReLU) via WMMA ----------------
// out[p, co] = relu?( scale[co]*sum_k im2col(in)[p,k]*W[k,co] + bias[co] )
// NOTE: grid is launched so that gridDim.x * MT == Npix exactly (all shapes in this net divide),
// so no pixel bound checks are needed.
template <int CIN, int COUT, bool RELU>
__global__ void conv3x3_wmma(const float* __restrict__ in, const unsigned short* __restrict__ wqT,
                             const float* __restrict__ scale, const float* __restrict__ bias,
                             float* __restrict__ out,
                             int H, int W, int Ho, int Wo, int stride, int pad) {
  constexpr int K  = 9 * CIN;
  constexpr int KP = ((K + 31) / 32) * 32;
  constexpr int nN = COUT / 16;      // wave tiles along Cout
  constexpr int nM = 8 / nN;         // wave tiles along pixels
  constexpr int MT = 16 * nM;        // pixels per workgroup

  __shared__ unsigned short As[MT * 32];     // [MT][32]  bf16 im2col tile
  __shared__ unsigned short Bs[COUT * 32];   // [COUT][32] bf16 weight tile (K-contiguous)

  const int tid  = threadIdx.x;
  const int lane = tid & 31;
  const int wv   = tid >> 5;
  const int tM   = wv / nN;
  const int tN   = wv - tM * nN;
  const int m0   = tM * 16;
  const int n0   = tN * 16;
  const int p0   = blockIdx.x * MT;
  const int HWo  = Ho * Wo;

  v8f acc = {};

  for (int k0 = 0; k0 < KP; k0 += 32) {
    __syncthreads();
    // ---- stage A (im2col, fp32 -> bf16) ----
    if constexpr (CIN % 4 == 0) {
      // vectorized: 4 consecutive k share one pixel & 4 contiguous channels (K%4==0, k0%4==0)
      for (int i = tid; i < MT * 8; i += 256) {
        int m  = i >> 3;
        int kq = (i & 7) << 2;
        int p  = p0 + m;
        int k  = k0 + kq;
        float4v val = {0.f, 0.f, 0.f, 0.f};
        if (k < K) {
          int n   = p / HWo;
          int rem = p - n * HWo;
          int yo  = rem / Wo;
          int xo  = rem - yo * Wo;
          int ky  = k / (3 * CIN);
          int r   = k - ky * (3 * CIN);
          int kx  = r / CIN;
          int ci  = r - kx * CIN;
          int yi  = yo * stride + ky - pad;
          int xi  = xo * stride + kx - pad;
          if (yi >= 0 && yi < H && xi >= 0 && xi < W)
            val = *(const float4v*)(in + (((size_t)n * H + yi) * W + xi) * CIN + ci);
        }
        ushort4v q;
        q.x = f2bf_u16(val.x);
        q.y = f2bf_u16(val.y);
        q.z = f2bf_u16(val.z);
        q.w = f2bf_u16(val.w);
        *(ushort4v*)(As + (m << 5) + kq) = q;
      }
    } else {
      for (int i = tid; i < MT * 32; i += 256) {
        int m  = i >> 5;
        int kk = i & 31;
        int p  = p0 + m;
        int k  = k0 + kk;
        float val = 0.f;
        if (k < K) {
          int n   = p / HWo;
          int rem = p - n * HWo;
          int yo  = rem / Wo;
          int xo  = rem - yo * Wo;
          int ky  = k / (3 * CIN);
          int r   = k - ky * (3 * CIN);
          int kx  = r / CIN;
          int ci  = r - kx * CIN;
          int yi  = yo * stride + ky - pad;
          int xi  = xo * stride + kx - pad;
          if (yi >= 0 && yi < H && xi >= 0 && xi < W)
            val = in[(((size_t)n * H + yi) * W + xi) * CIN + ci];
        }
        As[i] = f2bf_u16(val);
      }
    }
    // ---- stage B (weights, transposed layout: 32 contiguous halves per output channel) ----
    for (int i = tid; i < COUT * 4; i += 256) {
      int n  = i >> 2;
      int c8 = (i & 3) << 3;
      *(ushort8v*)(Bs + (n << 5) + c8) =
          *(const ushort8v*)(wqT + (size_t)n * KP + k0 + c8);
      if (k0 + 32 < KP)  // prefetch next K-step's weight chunk (global_prefetch_b8)
        __builtin_prefetch(wqT + (size_t)n * KP + k0 + 32 + c8, 0, 0);
    }
    __syncthreads();

    // A fragment: lane l<16 -> row m0+l, K {0..7,16..23}; lane l>=16 -> K {8..15,24..31}
    v16bf a, b;
    const int row = m0 + (lane & 15);
    const int kb  = (lane >> 4) * 8;
#pragma unroll
    for (int vv = 0; vv < 8; ++vv) {
      int kl = ((vv >= 4) ? 16 : 0) + kb + 2 * (vv & 3);
      a[2 * vv]     = bf_from_u16(As[(row << 5) + kl]);
      a[2 * vv + 1] = bf_from_u16(As[(row << 5) + kl + 1]);
    }
    // B fragment: lane l<16 -> col n0+l, K {0..7,16..23}; lane l>=16 -> K {8..15,24..31}
    const int col = n0 + (lane & 15);
    const int kb2 = (lane >> 4) * 8;
#pragma unroll
    for (int vv = 0; vv < 8; ++vv) {
      int kl2 = ((vv >= 4) ? 16 : 0) + kb2 + 2 * (vv & 3);
      b[2 * vv]     = bf_from_u16(Bs[(col << 5) + kl2]);
      b[2 * vv + 1] = bf_from_u16(Bs[(col << 5) + kl2 + 1]);
    }
    acc = __builtin_amdgcn_wmma_f32_16x16x32_bf16(false, a, false, b, (short)0, acc, false, false);
  }

  // epilogue: C/D layout: N = lane&15, M = v + (lane>=16 ? 8 : 0); no bound checks (exact tiling)
  const int col = n0 + (lane & 15);
  const float sc = scale[col];
  const float bs = bias[col];
  size_t pbase = (size_t)(p0 + m0 + ((lane >> 4) << 3)) * COUT + col;
#pragma unroll
  for (int vv = 0; vv < 8; ++vv) {
    float val = acc[vv] * sc + bs;
    if constexpr (RELU) val = fmaxf(val, 0.f);
    out[pbase + (size_t)vv * COUT] = val;
  }
}

// ---------------- elementwise / pooling / fc kernels ----------------
// downsample_b: avgpool2x2 then zero-pad channels C -> 2C
__global__ void downsample_kernel(const float* __restrict__ in, float* __restrict__ out,
                                  int total, int Ho, int Wo, int Cin) {
  int idx = blockIdx.x * blockDim.x + threadIdx.x;
  if (idx >= total) return;
  int C2 = 2 * Cin;
  int c = idx % C2;
  int r = idx / C2;
  int x = r % Wo; r /= Wo;
  int y = r % Ho;
  int n = r / Ho;
  float v = 0.f;
  if (c < Cin) {
    int W = 2 * Wo, H = 2 * Ho;
    const float* p = in + (((size_t)n * H + 2 * y) * W + 2 * x) * Cin + c;
    v = 0.25f * (p[0] + p[Cin] + p[(size_t)W * Cin] + p[(size_t)W * Cin + Cin]);
  }
  out[idx] = v;
}

// h = m ? relu(res + fx) : res   (policy==nullptr => always taken)
__global__ void combine_kernel(const float* __restrict__ res, const float* __restrict__ fx,
                               const float* __restrict__ policy, int bidx,
                               float* __restrict__ out, int total, int hwC) {
  int idx = blockIdx.x * blockDim.x + threadIdx.x;
  if (idx >= total) return;
  float m = 1.f;
  if (policy) {
    int n = idx / hwC;
    m = policy[n * 15 + bidx];
  }
  float r = res[idx];
  out[idx] = (m > 0.5f) ? fmaxf(r + fx[idx], 0.f) : r;
}

// global average pool: [B, HW, C] -> [B, C]
__global__ void gap_kernel(const float* __restrict__ in, float* __restrict__ feat, int HW, int C) {
  int n = blockIdx.x;
  int c = threadIdx.x;
  if (c >= C) return;
  float s = 0.f;
  for (int i = 0; i < HW; ++i) s += in[((size_t)n * HW + i) * C + c];
  feat[n * C + c] = s / (float)HW;
}

// z = feat @ w + b ; thresh: out = (sigmoid(z)>=0.5) == (z>=0)
__global__ void fc_kernel(const float* __restrict__ feat, const float* __restrict__ w,
                          const float* __restrict__ b, float* __restrict__ out,
                          int B, int Cf, int J, int thresh) {
  int idx = blockIdx.x * blockDim.x + threadIdx.x;
  if (idx >= B * J) return;
  int n = idx / J, j = idx - n * J;
  float z = b[j];
  for (int k = 0; k < Cf; ++k) z += feat[n * Cf + k] * w[k * J + j];
  out[idx] = thresh ? ((z >= 0.f) ? 1.f : 0.f) : z;
}

// ---------------- host orchestration ----------------
namespace {

struct ConvDesc {
  const float *w, *g, *b, *m, *v;
  int Cin, Cout;
  unsigned short* wq;
  float* sc;
  float* bi;
};

inline void launch_conv(const ConvDesc& c, const float* in, float* out,
                        int Hin, int stride, int relu, hipStream_t s) {
  int Ho = Hin / stride;
  int pad = (stride == 1) ? 1 : 0;  // jax SAME: s=1 -> (1,1); s=2, k=3, even in -> (0,1)
  int Npix = 512 * Ho * Ho;
#define CONV_CASE(CI, CO)                                                                  \
  if (c.Cin == CI && c.Cout == CO) {                                                       \
    constexpr int MT = 16 * (8 / (CO / 16));                                               \
    int grid = Npix / MT; /* exact for every shape in this network */                      \
    if (relu)                                                                              \
      conv3x3_wmma<CI, CO, true><<<grid, 256, 0, s>>>(in, c.wq, c.sc, c.bi, out,           \
                                                      Hin, Hin, Ho, Ho, stride, pad);      \
    else                                                                                   \
      conv3x3_wmma<CI, CO, false><<<grid, 256, 0, s>>>(in, c.wq, c.sc, c.bi, out,          \
                                                       Hin, Hin, Ho, Ho, stride, pad);     \
    return;                                                                                \
  }
  CONV_CASE(3, 16)
  CONV_CASE(16, 16)
  CONV_CASE(16, 32)
  CONV_CASE(32, 32)
  CONV_CASE(32, 64)
  CONV_CASE(64, 64)
#undef CONV_CASE
}

}  // namespace

extern "C" void kernel_launch(void* const* d_in, const int* in_sizes, int n_in,
                              void* d_out, int out_size, void* d_ws, size_t ws_size,
                              hipStream_t stream) {
  (void)in_sizes; (void)n_in; (void)out_size; (void)ws_size;
  const float* x = (const float*)d_in[0];

  // ----- workspace carve-up -----
  char* base = (char*)d_ws;
  size_t off = 0;
  auto take = [&](size_t bytes) -> void* {
    void* p = base + off;
    off = (off + bytes + 255) & ~(size_t)255;
    return p;
  };
  const size_t BIG = (size_t)512 * 32 * 32 * 16;  // max activation elems (fp32)
  float* bufs[4];
  for (int i = 0; i < 4; ++i) bufs[i] = (float*)take(BIG * 4);
  float* feat   = (float*)take((size_t)512 * 64 * 4);
  float* policy = (float*)take((size_t)512 * 15 * 4);

  // ----- conv descriptor table -----
  ConvDesc cv[38];
  int nc = 0;
  auto addconv = [&](int wi, int bni, int cin, int cout) -> int {
    ConvDesc c;
    c.w = (const float*)d_in[wi];
    c.g = (const float*)d_in[bni];
    c.b = (const float*)d_in[bni + 1];
    c.m = (const float*)d_in[bni + 2];
    c.v = (const float*)d_in[bni + 3];
    c.Cin = cin; c.Cout = cout;
    int K = 9 * cin, KP = ((K + 31) / 32) * 32;
    c.wq = (unsigned short*)take((size_t)KP * cout * 2);
    c.sc = (float*)take((size_t)cout * 4);
    c.bi = (float*)take((size_t)cout * 4);
    cv[nc] = c;
    return nc++;
  };

  // d_in layout (flattened dict order):
  // 0: x | 1: rnet seed_w | 2-5: seed_bn(g,b,m,v)
  // 6 + (si*5+bi)*10: {w1, bn1 g/b/m/v, w2, bn2 g/b/m/v} for s0b0..s2b4
  // 156: fc_w | 157: fc_b
  // 158: agent seed_w | 159-162: seed_bn | 163/173/183: agent b0/b1/b2 (same 10-entry layout)
  // 193: logit_w | 194: logit_b | 195: vnet_w | 196: vnet_b
  const int A_SEED = addconv(158, 159, 3, 16);
  const int acin[3] = {16, 16, 32}, acout[3] = {16, 32, 64};
  int AB[3][2];
  for (int k = 0; k < 3; ++k) {
    int bb = 163 + k * 10;
    AB[k][0] = addconv(bb, bb + 1, acin[k], acout[k]);
    AB[k][1] = addconv(bb + 5, bb + 6, acout[k], acout[k]);
  }
  const int R_SEED = addconv(1, 2, 3, 16);
  const int filt[3] = {16, 32, 64};
  int RB[3][5][2];
  {
    int prevC = 16;
    for (int si = 0; si < 3; ++si) {
      for (int bi = 0; bi < 5; ++bi) {
        int bb = 6 + (si * 5 + bi) * 10;
        int cin = (bi == 0) ? prevC : filt[si];
        RB[si][bi][0] = addconv(bb, bb + 1, cin, filt[si]);
        RB[si][bi][1] = addconv(bb + 5, bb + 6, filt[si], filt[si]);
      }
      prevC = filt[si];
    }
  }
  const float* fc_w    = (const float*)d_in[156];
  const float* fc_b    = (const float*)d_in[157];
  const float* logit_w = (const float*)d_in[193];
  const float* logit_b = (const float*)d_in[194];

  // ----- weight prep: bf16 transpose/convert + BN fold -----
  for (int i = 0; i < nc; ++i) {
    const ConvDesc& c = cv[i];
    int K = 9 * c.Cin, KP = ((K + 31) / 32) * 32;
    int tot = KP * c.Cout;
    cvt_w_kernel<<<(tot + 255) / 256, 256, 0, stream>>>(c.w, c.wq, K, KP, c.Cout);
    fold_bn_kernel<<<(c.Cout + 63) / 64, 64, 0, stream>>>(c.g, c.b, c.m, c.v, c.sc, c.bi, c.Cout);
  }

  const int EW = 256;
  auto ew_grid = [](int total) { return (total + 255) / 256; };

  // ================= agent forward =================
  launch_conv(cv[A_SEED], x, bufs[0], 32, 1, 1, stream);  // [512,32,32,16]
  // b0: 16->16, stride 1, identity residual
  launch_conv(cv[AB[0][0]], bufs[0], bufs[1], 32, 1, 1, stream);
  launch_conv(cv[AB[0][1]], bufs[1], bufs[2], 32, 1, 0, stream);
  {
    int total = 512 * 32 * 32 * 16;
    combine_kernel<<<ew_grid(total), EW, 0, stream>>>(bufs[0], bufs[2], nullptr, 0, bufs[1], total, 1);
  }
  // b1: 16->32, stride 2, downsample residual
  launch_conv(cv[AB[1][0]], bufs[1], bufs[2], 32, 2, 1, stream);  // [512,16,16,32]
  launch_conv(cv[AB[1][1]], bufs[2], bufs[3], 16, 1, 0, stream);
  {
    int total = 512 * 16 * 16 * 32;
    downsample_kernel<<<ew_grid(total), EW, 0, stream>>>(bufs[1], bufs[0], total, 16, 16, 16);
    combine_kernel<<<ew_grid(total), EW, 0, stream>>>(bufs[0], bufs[3], nullptr, 0, bufs[2], total, 1);
  }
  // b2: 32->64, stride 2, downsample residual
  launch_conv(cv[AB[2][0]], bufs[2], bufs[3], 16, 2, 1, stream);  // [512,8,8,64]
  launch_conv(cv[AB[2][1]], bufs[3], bufs[0], 8, 1, 0, stream);
  {
    int total = 512 * 8 * 8 * 64;
    downsample_kernel<<<ew_grid(total), EW, 0, stream>>>(bufs[2], bufs[1], total, 8, 8, 32);
    combine_kernel<<<ew_grid(total), EW, 0, stream>>>(bufs[1], bufs[0], nullptr, 0, bufs[3], total, 1);
  }
  gap_kernel<<<512, 64, 0, stream>>>(bufs[3], feat, 64, 64);
  fc_kernel<<<(512 * 15 + 255) / 256, 256, 0, stream>>>(feat, logit_w, logit_b, policy, 512, 64, 15, 1);

  // ================= rnet forward =================
  launch_conv(cv[R_SEED], x, bufs[0], 32, 1, 1, stream);  // [512,32,32,16]
  float* h = bufs[0];
  int curH = 32, curC = 16, gidx = 0;
  for (int si = 0; si < 3; ++si) {
    for (int bi = 0; bi < 5; ++bi) {
      int stride = (bi == 0 && si > 0) ? 2 : 1;
      int cout = filt[si];
      int Ho = curH / stride;
      float* f[3];
      int nf = 0;
      for (int q = 0; q < 4; ++q)
        if (bufs[q] != h) f[nf++] = bufs[q];
      launch_conv(cv[RB[si][bi][0]], h, f[0], curH, stride, 1, stream);
      launch_conv(cv[RB[si][bi][1]], f[0], f[1], Ho, 1, 0, stream);
      int total = 512 * Ho * Ho * cout;
      const float* res;
      if (bi == 0 && si > 0) {
        downsample_kernel<<<ew_grid(total), EW, 0, stream>>>(h, f[2], total, Ho, Ho, curC);
        res = f[2];
      } else {
        res = h;
      }
      combine_kernel<<<ew_grid(total), EW, 0, stream>>>(res, f[1], policy, gidx, f[0], total,
                                                        Ho * Ho * cout);
      h = f[0];
      curH = Ho;
      curC = cout;
      ++gidx;
    }
  }
  gap_kernel<<<512, 64, 0, stream>>>(h, feat, 64, 64);  // h: [512,8,8,64]
  fc_kernel<<<(512 * 10 + 255) / 256, 256, 0, stream>>>(feat, fc_w, fc_b, (float*)d_out,
                                                        512, 64, 10, 0);
}
